// GptOssAttentionLearn_28664611734205
// MI455X (gfx1250) — matrix operand order repaired and checked by reading, not compiled, not run on hardware
//
#include <hip/hip_runtime.h>
#include <hip/hip_bf16.h>
#include <stdint.h>

// ---------------------------------------------------------------------------
// GPT-OSS sliding-window attention layer for MI455X (gfx1250, wave32, WMMA).
//
// Shapes: B=1, S=2048, D=2048, H=32, KV=8, HD=64, SW=128, SCALE=1/8.
// Roofline: ~45 GFLOP vs ~72MB traffic -> all GEMMs through
// v_wmma_f32_16x16x32_bf16 (fp32 accumulate) near the 23.3 TB/s roofline.
// GEMM inner loop uses GLOBAL_LOAD_ASYNC_TO_LDS_B128 with a double-buffered
// LDS pipeline + s_wait_asynccnt (ASYNCcnt), the CDNA5 async data path.
// ---------------------------------------------------------------------------

typedef __attribute__((ext_vector_type(16))) __bf16        v16bf;
typedef __attribute__((ext_vector_type(8)))  float         v8f;
typedef __attribute__((ext_vector_type(8)))  unsigned int  v8u;
typedef unsigned short u16t;

#define S_LEN 2048
#define D_DIM 2048
#define NH    32
#define NKV   8
#define HD    64
#define SWIN  128
#define NEG_BIG (-3.0e38f)

#if defined(__AMDGCN__) && __has_builtin(__builtin_amdgcn_global_load_async_to_lds_b128) && __has_builtin(__builtin_amdgcn_s_wait_asynccnt)
#define HAVE_ASYNC_LDS 1
#else
#define HAVE_ASYNC_LDS 0
#endif

// ---- bf16 helpers (bit-level, round-to-nearest-even) ----------------------
static __device__ __forceinline__ unsigned int f2bfbits(float f) {
  unsigned int u = __builtin_bit_cast(unsigned int, f);
  unsigned int r = u + 0x7FFFu + ((u >> 16) & 1u);
  return r >> 16;
}
static __device__ __forceinline__ float bf2f(u16t h) {
  return __builtin_bit_cast(float, ((unsigned int)h) << 16);
}
static __device__ __forceinline__ unsigned int pack2bf(float lo, float hi) {
  return f2bfbits(lo) | (f2bfbits(hi) << 16);
}
static __device__ __forceinline__ v8u mk8(uint4 a, uint4 b) {
  return (v8u){a.x, a.y, a.z, a.w, b.x, b.y, b.z, b.w};
}
// D = A*B + C  (bf16 inputs held as packed uints, fp32 accumulator)
static __device__ __forceinline__ v8f wmma_bf16(v8u a, v8u b, v8f c) {
  return __builtin_amdgcn_wmma_f32_16x16x32_bf16(
      false, __builtin_bit_cast(v16bf, a),
      false, __builtin_bit_cast(v16bf, b),
      (short)0, c, false, false);
}

#if HAVE_ASYNC_LDS
// Builtin signature (from compiler diagnostic): (AS1 v4i* src, AS3 v4i* dst,
// imm offset, imm cpol) with v4i = gcc-style int vector_size(16).
typedef int v4i_t __attribute__((vector_size(16)));
typedef __attribute__((address_space(1))) v4i_t* gptr128_t;
typedef __attribute__((address_space(3))) v4i_t* lptr128_t;

// 16B async global->LDS copy. Generic LDS addresses carry the LDS offset in
// their low 32 bits (ISA 10.2 aperture mapping), so the int round-trip casts
// are safe and avoid ambiguous addrspace conversions.
static __device__ __forceinline__ void async_cp_b128(const u16t* g, u16t* l) {
  __builtin_amdgcn_global_load_async_to_lds_b128(
      (gptr128_t)(size_t)g,
      (lptr128_t)(unsigned int)(size_t)l,
      0, 0);
}
#endif

// ---- fp32 -> bf16 conversion ----------------------------------------------
__global__ void cvt_bf16_kernel(const float* __restrict__ in,
                                u16t* __restrict__ out, int n) {
  int i = blockIdx.x * 256 + threadIdx.x;
  if (i < n) out[i] = (u16t)f2bfbits(in[i]);
}

// ---- generic C = A (MxK, row-major) * B^T (B is NxK row-major) ------------
// Block: 256 threads = 8 waves. Block tile 128x128, K-step 32, double-
// buffered LDS stages filled by GLOBAL_LOAD_ASYNC_TO_LDS_B128 (4 async
// instructions per wave per stage), consumed after s_wait_asynccnt + barrier.
// LDS rows padded to 40 shorts (80B) so 16 lanes of 80B-strided b128 reads
// hit 16 distinct 16B bank groups (no 4-way conflict).
template <bool OUTF32>
__global__ __launch_bounds__(256)
void gemm_bt_kernel(const u16t* __restrict__ A, const u16t* __restrict__ B,
                    void* __restrict__ Cout, int M, int N, int K) {
  __shared__ u16t As[2][128 * 40];
  __shared__ u16t Bs[2][128 * 40];
  const int tid  = threadIdx.x;
  const int lane = tid & 31, wid = tid >> 5;
  const int half = lane >> 4, l15 = lane & 15;
  const int m0 = blockIdx.y * 128, n0 = blockIdx.x * 128;
  const int wm = (wid & 1) * 64, wn = (wid >> 1) * 32;
  const int lr = tid >> 1;          // 0..127 tile row for cooperative fill
  const int lc = (tid & 1) * 16;    // 0 or 16 (shorts)

  const u16t* gA = A + (size_t)(m0 + lr) * K + lc;  // + k0 per stage
  const u16t* gB = B + (size_t)(n0 + lr) * K + lc;

  v8f acc[4][2];
#pragma unroll
  for (int mi = 0; mi < 4; ++mi)
#pragma unroll
    for (int ni = 0; ni < 2; ++ni)
#pragma unroll
      for (int r = 0; r < 8; ++r) acc[mi][ni][r] = 0.0f;

  auto issue_stage = [&](int st, int k0) {
#if HAVE_ASYNC_LDS
    async_cp_b128(gA + k0,     &As[st][lr * 40 + lc]);
    async_cp_b128(gA + k0 + 8, &As[st][lr * 40 + lc + 8]);
    async_cp_b128(gB + k0,     &Bs[st][lr * 40 + lc]);
    async_cp_b128(gB + k0 + 8, &Bs[st][lr * 40 + lc + 8]);
#else
    const uint4* pa = (const uint4*)(gA + k0);
    const uint4* pb = (const uint4*)(gB + k0);
    uint4 a0 = pa[0], a1 = pa[1], b0 = pb[0], b1 = pb[1];
    *(uint4*)(&As[st][lr * 40 + lc])     = a0;
    *(uint4*)(&As[st][lr * 40 + lc + 8]) = a1;
    *(uint4*)(&Bs[st][lr * 40 + lc])     = b0;
    *(uint4*)(&Bs[st][lr * 40 + lc + 8]) = b1;
#endif
  };

  issue_stage(0, 0);               // prologue fill of stage 0
  int cur = 0;
  for (int k0 = 0; k0 < K; k0 += 32) {
    const bool more = (k0 + 32) < K;      // uniform across the block
    if (more) issue_stage(cur ^ 1, k0 + 32);
#if HAVE_ASYNC_LDS
    if (more) __builtin_amdgcn_s_wait_asynccnt(4);  // only next stage's 4 left
    else      __builtin_amdgcn_s_wait_asynccnt(0);
#endif
    __syncthreads();               // all waves' copies for `cur` visible

    // B-fragments: lane holds column n=l15, K contiguous 16*half..+15
    v8u bfr[2];
#pragma unroll
    for (int ni = 0; ni < 2; ++ni) {
      int n = wn + ni * 16 + l15;
      const uint4* p = (const uint4*)(&Bs[cur][n * 40 + half * 16]);
      bfr[ni] = mk8(p[0], p[1]);
    }
    // A-fragments: lane holds row m, K contiguous 16*half..+15
#pragma unroll
    for (int mi = 0; mi < 4; ++mi) {
      int r = wm + mi * 16 + l15;
      const uint4* p = (const uint4*)(&As[cur][r * 40 + half * 16]);
      v8u af = mk8(p[0], p[1]);  // +0 and +16 bytes
#pragma unroll
      for (int ni = 0; ni < 2; ++ni)
        acc[mi][ni] = wmma_bf16(af, bfr[ni], acc[mi][ni]);
    }
    __syncthreads();               // everyone done reading `cur` before reuse
    cur ^= 1;
  }

#pragma unroll
  for (int mi = 0; mi < 4; ++mi)
#pragma unroll
    for (int ni = 0; ni < 2; ++ni)
#pragma unroll
      for (int r = 0; r < 8; ++r) {
        int row = m0 + wm + mi * 16 + r + half * 8;  // C: M = r + 8*half
        int col = n0 + wn + ni * 16 + l15;           // C: N = lane&15
        float v = acc[mi][ni][r];
        if (OUTF32)
          ((float*)Cout)[(size_t)row * N + col] = v;
        else
          ((u16t*)Cout)[(size_t)row * N + col] = (u16t)f2bfbits(v);
      }
}

// ---- RoPE on bf16 Q/K (in place). One thread per (s, head, d<32) pair. ----
__global__ void rope_kernel(u16t* __restrict__ X, const float* __restrict__ cosT,
                            const float* __restrict__ sinT, int hshift) {
  int idx = blockIdx.x * 256 + threadIdx.x;
  int d = idx & 31;
  int t = idx >> 5;
  int nh = 1 << hshift;
  int hh = t & (nh - 1);
  int s  = t >> hshift;
  u16t* base = X + (((size_t)s << hshift) + hh) * (size_t)HD;
  float x1 = bf2f(base[d]);
  float x2 = bf2f(base[d + 32]);
  float c  = cosT[s * HD + d];   // emb=concat(freqs,freqs): cos[d]==cos[d+32]
  float si = sinT[s * HD + d];
  base[d]      = (u16t)f2bfbits(x1 * c - x2 * si);
  base[d + 32] = (u16t)f2bfbits(x2 * c + x1 * si);
}

// ---- V (s, kv*64+d) -> Vt[kv*64+d][s] so P.V B-operand loads contiguous ---
__global__ void vtrans_kernel(const u16t* __restrict__ V, u16t* __restrict__ Vt) {
  int idx = blockIdx.x * 256 + threadIdx.x;   // S*KV*HD = 1M
  int s = idx >> 9;
  int c = idx & 511;
  Vt[(size_t)c * S_LEN + s] = V[idx];
}

// ---- Sliding-window attention with sinks ----------------------------------
// One wave per (16-query tile, head). grid = (S/16, H/4), block = 128.
// S^T = K_tile * Q^T  (C layout: lane = query col -> per-lane softmax)
// O^T = Vt * P^T      (P B-frags built from exp'd S^T frags + one shfl_xor16)
__global__ __launch_bounds__(128)
void attn_kernel(const u16t* __restrict__ Qb, const u16t* __restrict__ Kb,
                 const u16t* __restrict__ Vt, const float* __restrict__ sinks,
                 u16t* __restrict__ attnB) {
  const int qt   = blockIdx.x;
  const int wid  = threadIdx.x >> 5;
  const int h    = blockIdx.y * 4 + wid;
  const int lane = threadIdx.x & 31;
  const int half = lane >> 4, l15 = lane & 15;
  const int q0   = qt * 16;
  const int hkv  = h >> 2;
  const int mcol = q0 + l15;      // the query column this lane owns

  // Q as two B-fragments (d = 0..31, 32..63): lane = query m, K contiguous.
  v8u qf[2];
  {
    const u16t* qrow = Qb + (size_t)mcol * D_DIM + h * HD;
#pragma unroll
    for (int f = 0; f < 2; ++f) {
      const uint4* p = (const uint4*)(qrow + f * 32 + half * 16);
      qf[f] = mk8(p[0], p[1]);
    }
  }

  // Scores S^T for up to 8 key tiles kt = qt-7 .. qt.
  float st[8][8];
  float lmax = NEG_BIG;
#pragma unroll
  for (int t = 0; t < 8; ++t)
#pragma unroll
    for (int r = 0; r < 8; ++r) st[t][r] = NEG_BIG;

#pragma unroll
  for (int t = 0; t < 8; ++t) {
    int kt = qt - 7 + t;
    if (kt < 0) continue;  // uniform across the wave
    const u16t* krow = Kb + (size_t)(kt * 16 + l15) * (NKV * HD) + hkv * HD;
    v8f s = {0.f, 0.f, 0.f, 0.f, 0.f, 0.f, 0.f, 0.f};
#pragma unroll
    for (int f = 0; f < 2; ++f) {
      const uint4* p = (const uint4*)(krow + f * 32 + half * 16);
      v8u kf = mk8(p[0], p[1]);          // A-frag: K contiguous 16*half..+15
      s = wmma_bf16(kf, qf[f], s);
    }
#pragma unroll
    for (int r = 0; r < 8; ++r) {
      int key = kt * 16 + r + half * 8;  // C layout: M(row)=key
      float v = s[r] * 0.125f;           // SCALE = HD^-0.5
      bool valid = (key <= mcol) && (mcol - key < SWIN);
      st[t][r] = valid ? v : NEG_BIG;
      lmax = fmaxf(lmax, st[t][r]);
    }
  }

  // Per-column softmax: merge the two half-waves, include the sink logit.
  float snk  = sinks[h];
  float rmax = fmaxf(fmaxf(lmax, __shfl_xor(lmax, 16, 32)), snk);
  float lsum = 0.0f;
#pragma unroll
  for (int t = 0; t < 8; ++t)
#pragma unroll
    for (int r = 0; r < 8; ++r) {
      float e = __expf(st[t][r] - rmax);
      st[t][r] = e;                      // P (unnormalized)
      lsum += e;
    }
  float denom = lsum + __shfl_xor(lsum, 16, 32) + __expf(snk - rmax);
  float inv = 1.0f / denom;

  // O^T = Vt * P^T over 4 key-tile pairs (32 keys each) x 4 d-tiles.
  v8f o[4];
#pragma unroll
  for (int dt = 0; dt < 4; ++dt)
#pragma unroll
    for (int r = 0; r < 8; ++r) o[dt][r] = 0.0f;

#pragma unroll
  for (int p = 0; p < 4; ++p) {
    int t = 2 * p, t1 = t + 1;
    // Build the P B-fragment: half-waves swap rows 8..15 / 0..7 via xor-16.
    float lo[8], hi[8];
#pragma unroll
    for (int r = 0; r < 8; ++r) {
      float send = (half == 0) ? st[t1][r] : st[t][r];
      float xv = __shfl_xor(send, 16, 32);
      lo[r] = (half == 0) ? st[t][r] : xv;   // K = base+0..7  (16..23 hi-half)
      hi[r] = (half == 0) ? xv : st[t1][r];  // K = base+8..15 (24..31 hi-half)
    }
    v8u pb;
#pragma unroll
    for (int j = 0; j < 4; ++j) {
      pb[j]     = pack2bf(lo[2 * j], lo[2 * j + 1]);
      pb[4 + j] = pack2bf(hi[2 * j], hi[2 * j + 1]);
    }
    int kbase = (qt - 7 + t) * 16;
#pragma unroll
    for (int dt = 0; dt < 4; ++dt) {
      const u16t* vrow = Vt + (size_t)(hkv * HD + dt * 16 + l15) * S_LEN;
      int c0 = kbase + half * 16;      if (c0 < 0) c0 = 0;  // P==0 there
      int c1 = kbase + half * 16 + 8;  if (c1 < 0) c1 = 0;
      v8u vf = mk8(*(const uint4*)(vrow + c0), *(const uint4*)(vrow + c1));
      o[dt] = wmma_bf16(vf, pb, o[dt]);
    }
  }

  // O^T C layout: lane = query m, rows = d. Normalize and store bf16.
#pragma unroll
  for (int dt = 0; dt < 4; ++dt)
#pragma unroll
    for (int r = 0; r < 8; ++r) {
      int d = dt * 16 + r + half * 8;
      attnB[(size_t)(q0 + l15) * D_DIM + h * HD + d] =
          (u16t)f2bfbits(o[dt][r] * inv);
    }
}

// ---------------------------------------------------------------------------
extern "C" void kernel_launch(void* const* d_in, const int* in_sizes, int n_in,
                              void* d_out, int out_size, void* d_ws, size_t ws_size,
                              hipStream_t stream) {
  const float* hidden = (const float*)d_in[0];
  const float* cosT   = (const float*)d_in[1];
  const float* sinT   = (const float*)d_in[2];
  const float* Wq     = (const float*)d_in[3];
  const float* Wk     = (const float*)d_in[4];
  const float* Wv     = (const float*)d_in[5];
  const float* Wo     = (const float*)d_in[6];
  const float* sinks  = (const float*)d_in[7];
  float* out = (float*)d_out;

  // Workspace layout (bf16 buffers), total ~50 MB.
  char* ws = (char*)d_ws;
  u16t* hB   = (u16t*)(ws);                       // 2048x2048
  u16t* WqB  = (u16t*)(ws + (8u << 20));          // 2048x2048
  u16t* WkB  = (u16t*)(ws + (16u << 20));         // 512x2048
  u16t* WvB  = (u16t*)(ws + (18u << 20));         // 512x2048
  u16t* WoB  = (u16t*)(ws + (20u << 20));         // 2048x2048
  u16t* Qb   = (u16t*)(ws + (28u << 20));         // 2048x2048
  u16t* Kb   = (u16t*)(ws + (36u << 20));         // 2048x512
  u16t* Vb   = (u16t*)(ws + (38u << 20));         // 2048x512
  u16t* Vtb  = (u16t*)(ws + (40u << 20));         // 512x2048
  u16t* attB = (u16t*)(ws + (42u << 20));         // 2048x2048

  // 1) fp32 -> bf16
  cvt_bf16_kernel<<<(S_LEN * D_DIM) / 256, 256, 0, stream>>>(hidden, hB, S_LEN * D_DIM);
  cvt_bf16_kernel<<<(NH * HD * D_DIM) / 256, 256, 0, stream>>>(Wq, WqB, NH * HD * D_DIM);
  cvt_bf16_kernel<<<(NKV * HD * D_DIM) / 256, 256, 0, stream>>>(Wk, WkB, NKV * HD * D_DIM);
  cvt_bf16_kernel<<<(NKV * HD * D_DIM) / 256, 256, 0, stream>>>(Wv, WvB, NKV * HD * D_DIM);
  cvt_bf16_kernel<<<(D_DIM * NH * HD) / 256, 256, 0, stream>>>(Wo, WoB, D_DIM * NH * HD);

  // 2) Projections: X @ W^T via WMMA
  gemm_bt_kernel<false><<<dim3(D_DIM / 128, S_LEN / 128), 256, 0, stream>>>(
      hB, WqB, Qb, S_LEN, D_DIM, D_DIM);
  gemm_bt_kernel<false><<<dim3((NKV * HD) / 128, S_LEN / 128), 256, 0, stream>>>(
      hB, WkB, Kb, S_LEN, NKV * HD, D_DIM);
  gemm_bt_kernel<false><<<dim3((NKV * HD) / 128, S_LEN / 128), 256, 0, stream>>>(
      hB, WvB, Vb, S_LEN, NKV * HD, D_DIM);

  // 3) RoPE (in place on bf16 Q, K)
  rope_kernel<<<(S_LEN * NH * 32) / 256, 256, 0, stream>>>(Qb, cosT, sinT, 5);
  rope_kernel<<<(S_LEN * NKV * 32) / 256, 256, 0, stream>>>(Kb, cosT, sinT, 3);

  // 4) V transpose for contiguous P.V fragments
  vtrans_kernel<<<(S_LEN * NKV * HD) / 256, 256, 0, stream>>>(Vb, Vtb);

  // 5) Sliding-window attention with sinks
  attn_kernel<<<dim3(S_LEN / 16, NH / 4), 128, 0, stream>>>(Qb, Kb, Vtb, sinks, attB);

  // 6) Output projection attn @ Wo^T -> fp32
  gemm_bt_kernel<true><<<dim3(D_DIM / 128, S_LEN / 128), 256, 0, stream>>>(
      attB, WoB, out, S_LEN, D_DIM, NH * HD);
}
